// TimeCycle_91096256348245
// MI455X (gfx1250) — matrix operand order repaired and checked by reading, not compiled, not run on hardware
//
#include <hip/hip_runtime.h>
#include <hip/hip_bf16.h>
#include <stdint.h>

// ---------------------------------------------------------------------------
// TimeCycle compute_affinity for MI455X (gfx1250, wave32, WMMA)
// B=2, C=128, N=4096 (H=W=64). Outputs (A, AA, log_AA, A1, A2), fp32.
// ---------------------------------------------------------------------------

typedef __attribute__((ext_vector_type(16))) _Float16 v16h;
typedef __attribute__((ext_vector_type(8)))  _Float16 v8h;
typedef __attribute__((ext_vector_type(8)))  float    v8f;

#define NDIM 4096
#define CDIM 128
#define BDIM 2
#define INV_T 14.285714285714286f   // 1/0.07
#define EPS_LOG 1e-20f

__device__ __forceinline__ v8f wmma_f16(v16h a, v16h b, v8f c) {
  // v_wmma_f32_16x16x32_f16  D = A*B + C (fp32 accumulate)
  return __builtin_amdgcn_wmma_f32_16x16x32_f16(
      /*neg_a=*/false, a, /*neg_b=*/false, b,
      /*c_mod=*/(short)0, c, /*reuse_a=*/false, /*reuse_b=*/false);
}

// A-matrix fragment (16x32 f16, row-major source, row = lane%16):
//   element i holds K = 8*(2*(i/8)+half)+(i%8)
//   -> two contiguous 16B chunks at K = kk+8*half and K = kk+16+8*half
__device__ __forceinline__ v16h load_afrag(const _Float16* __restrict__ row_ptr,
                                           int kk, int halfid) {
  union { v16h v; v8h h[2]; } u;
  const _Float16* p = row_ptr + kk + 8 * halfid;
  u.h[0] = *(const v8h*)(p);
  u.h[1] = *(const v8h*)(p + 16);
  return u.v;
}

// B-matrix fragment (32x16 f16, col = lane%16): element i holds K = 8*(2*(i/8)+half)+(i%8)
// -> two contiguous 16B chunks at K = kk+8*half and K = kk+16+8*half.
__device__ __forceinline__ v16h load_bfrag(const _Float16* __restrict__ col_ptr,
                                           int kk, int halfid) {
  union { v16h v; v8h h[2]; } u;
  const _Float16* p = col_ptr + kk + 8 * halfid;
  u.h[0] = *(const v8h*)(p);
  u.h[1] = *(const v8h*)(p + 16);
  return u.v;
}

// ---------------------------------------------------------------------------
// 1) L2-normalize over C and pack to f16, transposed to [B, N, C] row-major.
// ---------------------------------------------------------------------------
__global__ __launch_bounds__(256) void normalize_pack_kernel(
    const float* __restrict__ x, _Float16* __restrict__ xh) {
  const int b  = blockIdx.x;
  const int n  = blockIdx.y * 32 + threadIdx.x;
  const int ty = threadIdx.y;
  __shared__ float sm[8][32];

  const float* xb = x + (size_t)b * CDIM * NDIM;
  float s = 0.0f;
  for (int c = ty; c < CDIM; c += 8) {
    float v = xb[(size_t)c * NDIM + n];
    s += v * v;
  }
  sm[ty][threadIdx.x] = s;
  __syncthreads();
  if (ty == 0) {
    float t = sm[0][threadIdx.x];
    #pragma unroll
    for (int j = 1; j < 8; ++j) t += sm[j][threadIdx.x];
    sm[0][threadIdx.x] = 1.0f / fmaxf(sqrtf(t), 1e-12f);
  }
  __syncthreads();
  const float rinv = sm[0][threadIdx.x];

  _Float16* orow = xh + ((size_t)b * NDIM + n) * CDIM;
  for (int c = ty; c < CDIM; c += 8) {
    float v = xb[(size_t)c * NDIM + n];
    orow[c] = (_Float16)(v * rinv);
  }
}

// ---------------------------------------------------------------------------
// 2) GEMM1: A[b] = x1n^T * x2n  (N x N, K=128) via WMMA f16, mask+diag fused.
//    Block = 8 waves, tile 64x64; wave tile 16x32 (2 accumulators).
// ---------------------------------------------------------------------------
__global__ __launch_bounds__(256) void gemm1_corr_mask_kernel(
    const _Float16* __restrict__ x1h, const _Float16* __restrict__ x2h,
    float* __restrict__ A) {
  const int b      = blockIdx.z;
  const int m_blk  = blockIdx.x * 64;
  const int n_blk  = blockIdx.y * 64;
  const int w      = threadIdx.x >> 5;
  const int lane   = threadIdx.x & 31;
  const int halfid = lane >> 4;
  const int l16    = lane & 15;
  const int n0     = n_blk + 16 * (w >> 1);
  const int m0     = m_blk + 32 * (w & 1);

  const _Float16* x1b = x1h + (size_t)b * NDIM * CDIM;
  const _Float16* x2b = x2h + (size_t)b * NDIM * CDIM;
  const _Float16* arow  = x1b + (size_t)(n0 + l16) * CDIM;
  const _Float16* brow0 = x2b + (size_t)(m0 + l16) * CDIM;
  const _Float16* brow1 = x2b + (size_t)(m0 + 16 + l16) * CDIM;

  v8f acc0 = {}; v8f acc1 = {};
  #pragma unroll
  for (int kk = 0; kk < CDIM; kk += 32) {
    v16h af  = load_afrag(arow,  kk, halfid);
    v16h bf0 = load_bfrag(brow0, kk, halfid);
    v16h bf1 = load_bfrag(brow1, kk, halfid);
    acc0 = wmma_f16(af, bf0, acc0);
    acc1 = wmma_f16(af, bf1, acc1);
  }

  float* Ab = A + (size_t)b * NDIM * NDIM;
  #pragma unroll
  for (int r = 0; r < 8; ++r) {
    const int n  = n0 + r + 8 * halfid;       // C/D layout: M = vgpr + 8*half
    const int gx = n >> 6, gy = n & 63;
    {
      const int m  = m0 + l16;                // N = lane%16
      const int dx = gx - (m >> 6), dy = gy - (m & 63);
      const bool ok = (dx * dx + dy * dy < 100) && (n != m);
      Ab[(size_t)n * NDIM + m] = ok ? acc0[r] : 0.0f;
    }
    {
      const int m  = m0 + 16 + l16;
      const int dx = gx - (m >> 6), dy = gy - (m & 63);
      const bool ok = (dx * dx + dy * dy < 100) && (n != m);
      Ab[(size_t)n * NDIM + m] = ok ? acc1[r] : 0.0f;
    }
  }
}

// ---------------------------------------------------------------------------
// 3) Row stats: per (b,n): max over row, then sum of exp((a-max)/T).
// ---------------------------------------------------------------------------
__global__ __launch_bounds__(256) void rowstats_kernel(
    const float* __restrict__ A, float* __restrict__ rowmax,
    float* __restrict__ rowsum) {
  const int row = blockIdx.x;               // 0 .. B*N-1
  const int tid = threadIdx.x;
  const float* Ar = A + (size_t)row * NDIM;
  __shared__ float sm[256];

  float mx = -3.0e38f;
  for (int m = tid; m < NDIM; m += 256) mx = fmaxf(mx, Ar[m]);
  sm[tid] = mx; __syncthreads();
  for (int s = 128; s > 0; s >>= 1) {
    if (tid < s) sm[tid] = fmaxf(sm[tid], sm[tid + s]);
    __syncthreads();
  }
  const float rmax = sm[0];
  __syncthreads();

  float ss = 0.0f;
  for (int m = tid; m < NDIM; m += 256) ss += __expf((Ar[m] - rmax) * INV_T);
  sm[tid] = ss; __syncthreads();
  for (int s = 128; s > 0; s >>= 1) {
    if (tid < s) sm[tid] += sm[tid + s];
    __syncthreads();
  }
  if (tid == 0) { rowmax[row] = rmax; rowsum[row] = sm[0]; }
}

// ---------------------------------------------------------------------------
// 4) Column stats (for A2 = softmax of A^T rows): coalesced tiled column scan.
// ---------------------------------------------------------------------------
__global__ __launch_bounds__(256) void colstats_kernel(
    const float* __restrict__ A, float* __restrict__ colmax,
    float* __restrict__ colsum) {
  const int b = blockIdx.x;
  const int m = blockIdx.y * 32 + threadIdx.x;
  const int ty = threadIdx.y;
  const float* Ab = A + (size_t)b * NDIM * NDIM;
  __shared__ float sm[8][32];

  float mx = -3.0e38f;
  for (int n = ty; n < NDIM; n += 8) mx = fmaxf(mx, Ab[(size_t)n * NDIM + m]);
  sm[ty][threadIdx.x] = mx; __syncthreads();
  if (ty == 0) {
    #pragma unroll
    for (int j = 1; j < 8; ++j) mx = fmaxf(mx, sm[j][threadIdx.x]);
    sm[0][threadIdx.x] = mx;
  }
  __syncthreads();
  mx = sm[0][threadIdx.x];
  __syncthreads();

  float ss = 0.0f;
  for (int n = ty; n < NDIM; n += 8)
    ss += __expf((Ab[(size_t)n * NDIM + m] - mx) * INV_T);
  sm[ty][threadIdx.x] = ss; __syncthreads();
  if (ty == 0) {
    #pragma unroll
    for (int j = 1; j < 8; ++j) ss += sm[j][threadIdx.x];
    colmax[b * NDIM + m] = mx;
    colsum[b * NDIM + m] = ss;
  }
}

// ---------------------------------------------------------------------------
// 5) Write A1, A2 (fp32 outputs) + f16 copies for GEMM2, transposing through
//    LDS so A2h is row-major [n,m] of A2 and A1hT is [c][k].
// ---------------------------------------------------------------------------
__global__ __launch_bounds__(256) void softmax_write_kernel(
    const float* __restrict__ A,
    const float* __restrict__ rowmax, const float* __restrict__ rowsum,
    const float* __restrict__ colmax, const float* __restrict__ colsum,
    float* __restrict__ A1, float* __restrict__ A2,
    _Float16* __restrict__ A1hT, _Float16* __restrict__ A2h) {
  const int b  = blockIdx.z;
  const int m0 = blockIdx.x * 32;
  const int n0 = blockIdx.y * 32;
  const int tx = threadIdx.x, ty = threadIdx.y;
  __shared__ float e1s[32][33];
  __shared__ float e2s[32][33];

  const size_t NN = (size_t)NDIM * NDIM;
  const float* Ab  = A  + (size_t)b * NN;
  float* A1b = A1 + (size_t)b * NN;
  float* A2b = A2 + (size_t)b * NN;
  _Float16* A1Tb = A1hT + (size_t)b * NN;
  _Float16* A2hb = A2h  + (size_t)b * NN;

  const int m = m0 + tx;
  const float cmax = colmax[b * NDIM + m];
  const float cinv = 1.0f / colsum[b * NDIM + m];

  #pragma unroll
  for (int j = 0; j < 4; ++j) {
    const int nl = ty + 8 * j;
    const int n  = n0 + nl;
    const float a    = Ab[(size_t)n * NDIM + m];
    const float rmax = rowmax[b * NDIM + n];
    const float rinv = 1.0f / rowsum[b * NDIM + n];
    const float e1 = __expf((a - rmax) * INV_T) * rinv;  // A1[n][m]
    const float e2 = __expf((a - cmax) * INV_T) * cinv;  // A2[m][n]
    A1b[(size_t)n * NDIM + m] = e1;
    e1s[tx][nl] = e1;     // [mLocal][nLocal]
    e2s[tx][nl] = e2;
  }
  __syncthreads();
  #pragma unroll
  for (int j = 0; j < 4; ++j) {
    const int ml = ty + 8 * j;
    const int mg = m0 + ml;
    const int ng = n0 + tx;
    const float e1 = e1s[ml][tx];
    const float e2 = e2s[ml][tx];
    A2b[(size_t)mg * NDIM + ng]  = e2;             // coalesced over tx
    A2hb[(size_t)mg * NDIM + ng] = (_Float16)e2;   // A2 row-major
    A1Tb[(size_t)mg * NDIM + ng] = (_Float16)e1;   // A1 transposed [c][k]
  }
}

// ---------------------------------------------------------------------------
// 6) GEMM2: AA = A2 * A1 (4096^3 per batch) via WMMA f16 / fp32 accumulate.
//    Block = 8 waves, tile 128x128; wave tile 32x64 (8 accumulators).
//    8 WMMA per 6 fragment loads per k-step (~21 flop/L2-byte per lane);
//    A2h/A1hT (67MB/batch pair) stay resident in the 192MB L2.
// ---------------------------------------------------------------------------
__global__ __launch_bounds__(256) void gemm2_AA_kernel(
    const _Float16* __restrict__ A2h, const _Float16* __restrict__ A1hT,
    float* __restrict__ AA, float* __restrict__ logAA) {
  const int b      = blockIdx.z;
  const int c_blk  = blockIdx.x * 128;
  const int n_blk  = blockIdx.y * 128;
  const int w      = threadIdx.x >> 5;
  const int lane   = threadIdx.x & 31;
  const int halfid = lane >> 4;
  const int l16    = lane & 15;
  const int n0     = n_blk + 32 * (w & 3);   // 4 row groups of 32 -> 128 rows
  const int c0     = c_blk + 64 * (w >> 2);  // 2 col groups of 64 -> 128 cols

  const size_t NN = (size_t)NDIM * NDIM;
  const _Float16* a2b = A2h  + (size_t)b * NN;
  const _Float16* a1b = A1hT + (size_t)b * NN;
  const _Float16* ar0 = a2b + (size_t)(n0 + l16) * NDIM;
  const _Float16* ar1 = a2b + (size_t)(n0 + 16 + l16) * NDIM;
  const _Float16* br0 = a1b + (size_t)(c0 + l16) * NDIM;
  const _Float16* br1 = a1b + (size_t)(c0 + 16 + l16) * NDIM;
  const _Float16* br2 = a1b + (size_t)(c0 + 32 + l16) * NDIM;
  const _Float16* br3 = a1b + (size_t)(c0 + 48 + l16) * NDIM;

  v8f acc[2][4];
  #pragma unroll
  for (int i = 0; i < 2; ++i)
    #pragma unroll
    for (int j = 0; j < 4; ++j) acc[i][j] = (v8f){};

  for (int kk = 0; kk < NDIM; kk += 32) {
    // keep the L2->WGP stream ahead of the WMMA pipe
    __builtin_prefetch(ar0 + kk + 128, 0, 1);
    __builtin_prefetch(br0 + kk + 128, 0, 1);
    v16h a0 = load_afrag(ar0, kk, halfid);
    v16h a1 = load_afrag(ar1, kk, halfid);
    v16h b0 = load_bfrag(br0, kk, halfid);
    v16h b1 = load_bfrag(br1, kk, halfid);
    v16h b2 = load_bfrag(br2, kk, halfid);
    v16h b3 = load_bfrag(br3, kk, halfid);
    acc[0][0] = wmma_f16(a0, b0, acc[0][0]);
    acc[0][1] = wmma_f16(a0, b1, acc[0][1]);
    acc[0][2] = wmma_f16(a0, b2, acc[0][2]);
    acc[0][3] = wmma_f16(a0, b3, acc[0][3]);
    acc[1][0] = wmma_f16(a1, b0, acc[1][0]);
    acc[1][1] = wmma_f16(a1, b1, acc[1][1]);
    acc[1][2] = wmma_f16(a1, b2, acc[1][2]);
    acc[1][3] = wmma_f16(a1, b3, acc[1][3]);
  }

  float* AAb = AA    + (size_t)b * NN;
  float* Lb  = logAA + (size_t)b * NN;
  #pragma unroll
  for (int i = 0; i < 2; ++i) {
    #pragma unroll
    for (int j = 0; j < 4; ++j) {
      const int cg = c0 + 16 * j + l16;       // C/D layout: N = lane%16
      #pragma unroll
      for (int r = 0; r < 8; ++r) {
        const int ng = n0 + 16 * i + r + 8 * halfid;  // M = vgpr + 8*half
        const float v = acc[i][j][r];
        const size_t idx = (size_t)ng * NDIM + cg;
        AAb[idx] = v;
        Lb[idx]  = __logf(v + EPS_LOG);
      }
    }
  }
}

// ---------------------------------------------------------------------------
// Host-side orchestration
// ---------------------------------------------------------------------------
extern "C" void kernel_launch(void* const* d_in, const int* in_sizes, int n_in,
                              void* d_out, int out_size, void* d_ws, size_t ws_size,
                              hipStream_t stream) {
  (void)in_sizes; (void)n_in; (void)out_size; (void)ws_size;
  const float* x1 = (const float*)d_in[0];
  const float* x2 = (const float*)d_in[1];

  const size_t NN = (size_t)NDIM * NDIM;
  const size_t S  = (size_t)BDIM * NN;           // elements per output tensor
  float* out  = (float*)d_out;
  float* A    = out;            // output 0
  float* AA   = out + S;        // output 1
  float* logA = out + 2 * S;    // output 2
  float* A1   = out + 3 * S;    // output 3
  float* A2   = out + 4 * S;    // output 4

  // Workspace carve-up (256B aligned): ~138.4 MB total
  uint8_t* ws = (uint8_t*)d_ws;
  size_t off = 0;
  auto carve = [&](size_t bytes) -> void* {
    void* p = ws + off;
    off += (bytes + 255) & ~(size_t)255;
    return p;
  };
  _Float16* x1h  = (_Float16*)carve((size_t)BDIM * NDIM * CDIM * sizeof(_Float16));
  _Float16* x2h  = (_Float16*)carve((size_t)BDIM * NDIM * CDIM * sizeof(_Float16));
  float* rowmax  = (float*)carve((size_t)BDIM * NDIM * sizeof(float));
  float* rowsum  = (float*)carve((size_t)BDIM * NDIM * sizeof(float));
  float* colmax  = (float*)carve((size_t)BDIM * NDIM * sizeof(float));
  float* colsum  = (float*)carve((size_t)BDIM * NDIM * sizeof(float));
  _Float16* A1hT = (_Float16*)carve(S * sizeof(_Float16));
  _Float16* A2h  = (_Float16*)carve(S * sizeof(_Float16));

  // 1) normalize + pack f16 [B,N,C]
  {
    dim3 g(BDIM, NDIM / 32), blk(32, 8);
    normalize_pack_kernel<<<g, blk, 0, stream>>>(x1, x1h);
    normalize_pack_kernel<<<g, blk, 0, stream>>>(x2, x2h);
  }
  // 2) masked correlation A (WMMA)
  {
    dim3 g(NDIM / 64, NDIM / 64, BDIM);
    gemm1_corr_mask_kernel<<<g, 256, 0, stream>>>(x1h, x2h, A);
  }
  // 3) row stats
  rowstats_kernel<<<BDIM * NDIM, 256, 0, stream>>>(A, rowmax, rowsum);
  // 4) col stats
  {
    dim3 g(BDIM, NDIM / 32), blk(32, 8);
    colstats_kernel<<<g, blk, 0, stream>>>(A, colmax, colsum);
  }
  // 5) A1 / A2 + f16 copies (fused transpose)
  {
    dim3 g(NDIM / 32, NDIM / 32, BDIM), blk(32, 8);
    softmax_write_kernel<<<g, blk, 0, stream>>>(A, rowmax, rowsum, colmax,
                                                colsum, A1, A2, A1hT, A2h);
  }
  // 6) AA = A2*A1 + log epilogue (WMMA, dominant cost)
  {
    dim3 g(NDIM / 128, NDIM / 128, BDIM);
    gemm2_AA_kernel<<<g, 256, 0, stream>>>(A2h, A1hT, AA, logA);
  }
}
